// Model_56564719288437
// MI455X (gfx1250) — matrix operand order repaired and checked by reading, not compiled, not run on hardware
//
#include <hip/hip_runtime.h>

#define SEQ   2048
#define BATCH 2
#define NSB   (SEQ * BATCH)   // 4096 independent problems
#define RATE  4
#define HID   3584            // 14 * 256

typedef __attribute__((ext_vector_type(16))) __bf16 v16bf;
typedef __attribute__((ext_vector_type(8)))  __bf16 v8bf;
typedef __attribute__((ext_vector_type(8)))  float  v8f;

// One block per (s,b). Fully fused backward:
//   grad_h_res[r][q]  = sum_h g[r,h] * x[q,h]          (WMMA bf16 -> f32)
//   grad_h_post[r]    = sum_h g[r,h] * sub[h]          (VALU + wave/LDS reduce)
//   grad_x[q][h]      = sum_r h_res[r][q] * g[r,h]     (VALU, streamed)
//   grad_sub[h]       = sum_r g[r,h] * h_post[r]       (VALU, streamed)
__global__ __launch_bounds__(256) void fused_stream_bwd(
    const __bf16* __restrict__ h_res,   // [NSB,4,4]
    const __bf16* __restrict__ h_post,  // [NSB,4]
    const __bf16* __restrict__ xs,      // [NSB,4,HID]
    const __bf16* __restrict__ sub,     // [NSB,HID]
    const __bf16* __restrict__ g,       // [NSB,4,HID]
    __bf16* __restrict__ o_hres,        // [NSB,16]
    __bf16* __restrict__ o_hpost,       // [NSB,4]
    __bf16* __restrict__ o_gx,          // [NSB,4,HID]
    __bf16* __restrict__ o_gsub)        // [NSB,HID]
{
    const int sb   = blockIdx.x;
    const int tid  = threadIdx.x;
    const int lane = tid & 31;
    const int wv   = tid >> 5;

    const __bf16* g0 = g   + (size_t)sb * (RATE * HID);
    const __bf16* x0 = xs  + (size_t)sb * (RATE * HID);
    const __bf16* s0 = sub + (size_t)sb * HID;
    __bf16* ogx = o_gx   + (size_t)sb * (RATE * HID);
    __bf16* ogs = o_gsub + (size_t)sb * HID;

    // Per-problem scalars (tiny, L0-resident)
    float hrf[RATE][RATE];
    float hpf[RATE];
    {
        const __bf16* hr = h_res + (size_t)sb * 16;
        const __bf16* hp = h_post + (size_t)sb * 4;
#pragma unroll
        for (int r = 0; r < RATE; ++r) {
            hpf[r] = (float)hp[r];
#pragma unroll
            for (int q = 0; q < RATE; ++q) hrf[r][q] = (float)hr[r * 4 + q];
        }
    }

    // WMMA fragment addressing (16-bit A 16x32 / B 32x16; contiguous k = 16h + s, same for A and B).
    // Rows/cols >= RATE are don't-care for D[r<4][q<4]; clamp with &3 to keep
    // addresses in-bounds (redundant loads hit cache).
    const int m_eff  = lane & 3;                 // A row / B col this lane feeds
    const int a_koff = (lane < 16) ? 0 : 16;     // A: low half K=0-15, high K=16-31
    const int b_koff = (lane < 16) ? 0 : 16;     // B: low half K=0-15, high K=16-31

    v8f   acc = {};                              // grad_h_res accumulator (16x16 f32)
    float hp_acc[RATE] = {0.f, 0.f, 0.f, 0.f};   // grad_h_post partials

    for (int c = wv; c < HID / 256; c += 8) {
        const int base = c * 256;
        const int h0   = base + lane * 8;

        // ---- streamed elementwise pass (coalesced b128 loads/stores) ----
        v8bf gv[RATE];
#pragma unroll
        for (int r = 0; r < RATE; ++r)
            gv[r] = *(const v8bf*)(g0 + r * HID + h0);
        v8bf sv = *(const v8bf*)(s0 + h0);

        v8bf osub;
        v8bf oq[RATE];
#pragma unroll
        for (int i = 0; i < 8; ++i) {
            float gf[RATE];
#pragma unroll
            for (int r = 0; r < RATE; ++r) gf[r] = (float)gv[r][i];
            const float sf = (float)sv[i];

            osub[i] = (__bf16)(gf[0] * hpf[0] + gf[1] * hpf[1] +
                               gf[2] * hpf[2] + gf[3] * hpf[3]);
#pragma unroll
            for (int q = 0; q < RATE; ++q)
                oq[q][i] = (__bf16)(hrf[0][q] * gf[0] + hrf[1][q] * gf[1] +
                                    hrf[2][q] * gf[2] + hrf[3][q] * gf[3]);
#pragma unroll
            for (int r = 0; r < RATE; ++r) hp_acc[r] += gf[r] * sf;
        }
        *(v8bf*)(ogs + h0) = osub;
#pragma unroll
        for (int q = 0; q < RATE; ++q)
            *(v8bf*)(ogx + q * HID + h0) = oq[q];

        // ---- WMMA pass: grad_h_res += g(16x32) * x^T(32x16), K=32 tiles ----
#pragma unroll
        for (int k = 0; k < 8; ++k) {
            const int hb = base + k * 32;
            // A fragment: g rows (cache-hot from the pass above)
            v8bf a_lo = *(const v8bf*)(g0 + m_eff * HID + hb + a_koff);
            v8bf a_hi = *(const v8bf*)(g0 + m_eff * HID + hb + 8 + a_koff);
            v16bf A;
#pragma unroll
            for (int i = 0; i < 8; ++i) { A[i] = a_lo[i]; A[i + 8] = a_hi[i]; }
            // B fragment: x rows, 16 contiguous bf16 per lane
            v16bf B = *(const v16bf*)(x0 + m_eff * HID + hb + b_koff);

            acc = __builtin_amdgcn_wmma_f32_16x16x32_bf16(
                false, A, false, B, (short)0, acc, false, false);
        }
    }

    // ---- cross-wave reductions ----
    __shared__ float lds_hres[8][16];
    __shared__ float lds_hpost[8][4];

    // D[r][q] sits in acc[r] of lane q (C/D layout: VGPR v -> M=v, lane -> N)
    if (lane < 4) {
#pragma unroll
        for (int r = 0; r < RATE; ++r) lds_hres[wv][r * 4 + lane] = acc[r];
    }
#pragma unroll
    for (int off = 16; off >= 1; off >>= 1) {
#pragma unroll
        for (int r = 0; r < RATE; ++r)
            hp_acc[r] += __shfl_xor(hp_acc[r], off, 32);
    }
    if (lane == 0) {
#pragma unroll
        for (int r = 0; r < RATE; ++r) lds_hpost[wv][r] = hp_acc[r];
    }
    __syncthreads();

    if (tid < 16) {
        float s = 0.f;
#pragma unroll
        for (int w = 0; w < 8; ++w) s += lds_hres[w][tid];
        o_hres[(size_t)sb * 16 + tid] = (__bf16)s;
    }
    if (tid < 4) {
        float s = 0.f;
#pragma unroll
        for (int w = 0; w < 8; ++w) s += lds_hpost[w][tid];
        o_hpost[(size_t)sb * 4 + tid] = (__bf16)s;
    }
}

extern "C" void kernel_launch(void* const* d_in, const int* in_sizes, int n_in,
                              void* d_out, int out_size, void* d_ws, size_t ws_size,
                              hipStream_t stream) {
    const __bf16* h_res  = (const __bf16*)d_in[0];
    const __bf16* h_post = (const __bf16*)d_in[1];
    const __bf16* xs     = (const __bf16*)d_in[2];
    const __bf16* sub    = (const __bf16*)d_in[3];
    const __bf16* g      = (const __bf16*)d_in[4];

    __bf16* out     = (__bf16*)d_out;
    __bf16* o_hres  = out;                                  // [NSB,4,4]
    __bf16* o_hpost = o_hres + (size_t)NSB * 16;            // [NSB,4,1]
    __bf16* o_gx    = o_hpost + (size_t)NSB * 4;            // [NSB,RATE*HID]
    __bf16* o_gsub  = o_gx + (size_t)NSB * RATE * HID;      // [NSB,HID]

    fused_stream_bwd<<<dim3(NSB), dim3(256), 0, stream>>>(
        h_res, h_post, xs, sub, g, o_hres, o_hpost, o_gx, o_gsub);
}
